// TransitionPathE3NNLayer_31421980737958
// MI455X (gfx1250) — matrix operand order repaired and checked by reading, not compiled, not run on hardware
//
#include <hip/hip_runtime.h>
#include <hip/hip_bf16.h>
#include <math.h>

typedef __attribute__((ext_vector_type(16))) _Float16 v16h;
typedef __attribute__((ext_vector_type(8)))  _Float16 v8h;
typedef __attribute__((ext_vector_type(8)))  float    v8f;

#define NESF 37           // 7 + 3*NRBF edge scalar features
#define WNUM 576          // 256 + 128 + 128 + 64
#define NRBF 10
#define CUTV 5.0f
#define GAMMA 3.24f       // 1/(CUT/(NRBF-1))^2
#define SQ3C 1.7320508075688772f
#define INV_FAN 0.20412414523193154f  // 1/sqrt(24)

static __host__ int imin_h(int a, int b){ return a < b ? a : b; }

// tanh-GELU == x * sigmoid(2*sqrt(2/pi)*(x + 0.044715 x^3)); branch-free:
// one native v_exp_f32 + one v_rcp_f32.
__device__ __forceinline__ float gelu_t(float v){
  const float k2 = 1.5957691216057308f;     // 2*sqrt(2/pi)
  float z = k2 * (v + 0.044715f * v * v * v);
  float t = __builtin_amdgcn_rcpf(1.0f + __expf(-z));
  return v * t;
}

// ---- WMMA fragment helpers (gfx1250 wave32 layouts per ISA 7.12.2) ----
__device__ __forceinline__ v16h load_a_frag(const _Float16* buf, int lda, int kOff){
  int lane = threadIdx.x & 31;
  int row  = lane & 15;
  int kb   = (lane >> 4) << 3;                 // 0 or 8
  const _Float16* p0 = buf + row * lda + kOff + kb;        // K = kb..kb+7
  v8h lo = *(const v8h*)(p0);                              // 16B aligned
  v8h hi = *(const v8h*)(p0 + 16);                         // K = 16+kb..
  v16h a;
  #pragma unroll
  for (int j = 0; j < 8; ++j){ a[j] = lo[j]; a[8 + j] = hi[j]; }
  return a;
}
// Weights stored K-major transposed: wT[n*ldk + k]
__device__ __forceinline__ v16h load_b_frag(const _Float16* wT, int ldk, int nOff, int kOff){
  int lane = threadIdx.x & 31;
  int col  = lane & 15;
  int kb   = (lane >> 4) << 3;                 // 0 or 8
  const _Float16* p = wT + (nOff + col) * ldk + kOff + kb;
  v8h lo = *(const v8h*)(p);
  v8h hi = *(const v8h*)(p + 16);
  v16h b;
  #pragma unroll
  for (int j = 0; j < 8; ++j){ b[j] = lo[j]; b[8 + j] = hi[j]; }
  return b;
}
__device__ __forceinline__ v8f wmma_f16(v16h a, v16h b, v8f c){
  return __builtin_amdgcn_wmma_f32_16x16x32_f16(false, a, false, b, (short)0, c, false, false);
}

// ---- edge scalar features (37) ----
__device__ __forceinline__ void compute_es(_Float16* row,
    const float* __restrict__ x, const float* __restrict__ xA, const float* __restrict__ xB,
    int is, int id, int bA, int bB, float* dirOut)
{
  float ex = x[id*3+0]-x[is*3+0], ey = x[id*3+1]-x[is*3+1], ez = x[id*3+2]-x[is*3+2];
  float d2 = fmaxf(ex*ex+ey*ey+ez*ez, 1e-8f);
  float rinv = __builtin_amdgcn_rsqf(d2);     // v_rsq_f32
  float dist = d2 * rinv;                     // sqrt(d2)
  dirOut[0] = ex*rinv; dirOut[1] = ey*rinv; dirOut[2] = ez*rinv;
  float ax = xA[id*3+0]-xA[is*3+0], ay = xA[id*3+1]-xA[is*3+1], az = xA[id*3+2]-xA[is*3+2];
  float dA = __builtin_amdgcn_sqrtf(fmaxf(ax*ax+ay*ay+az*az, 1e-8f));
  float bx = xB[id*3+0]-xB[is*3+0], by = xB[id*3+1]-xB[is*3+1], bz = xB[id*3+2]-xB[is*3+2];
  float dB = __builtin_amdgcn_sqrtf(fmaxf(bx*bx+by*by+bz*bz, 1e-8f));
  row[0] = (_Float16)(float)bA;  row[1] = (_Float16)(float)bB;
  row[2] = (_Float16)dist;       row[3] = (_Float16)d2;
  row[4] = (_Float16)dA;         row[5] = (_Float16)dB;
  row[6] = (_Float16)(dA - dB);
  const float step = CUTV / (NRBF - 1);
  #pragma unroll
  for (int r = 0; r < NRBF; ++r){
    float mu = r * step;
    float t0 = dist - mu, t1 = dA - mu, t2 = dB - mu;
    row[7  + r] = (_Float16)__expf(-GAMMA * t0 * t0);
    row[17 + r] = (_Float16)__expf(-GAMMA * t1 * t1);
    row[27 + r] = (_Float16)__expf(-GAMMA * t2 * t2);
  }
}

// ================= Pass 1: r-MLP + tensor-product messages + scatter ============
struct P1Shared {
  alignas(16) _Float16 Wr1t[64*64];     // [n][k], k>=37 zero-padded
  alignas(16) _Float16 Wr2t[64*64];
  alignas(16) _Float16 Wr3t[WNUM*64];   // 73728 B
  float rb1s[64], rb2s[64], rb3s[WNUM];
  struct {
    alignas(16) _Float16 es[16*64];     // edge scalars -> later h2
    alignas(16) _Float16 h[16*64];      // h1
    float f0s[16*16];
    float f1s[16*24];
    float pdir[16*8];       // sum_m f1s[u][m]*dir[m]
    float dir[16*3];
    int   dste[16];
  } wv[4];
};

__global__ __launch_bounds__(128) void edge_pass1(
    const float* __restrict__ f, const float* __restrict__ x,
    const float* __restrict__ xA, const float* __restrict__ xB,
    const float* __restrict__ rW1, const float* __restrict__ rb1,
    const float* __restrict__ rW2, const float* __restrict__ rb2,
    const float* __restrict__ rW3, const float* __restrict__ rb3,
    const int* __restrict__ src, const int* __restrict__ dst,
    const int* __restrict__ bondA, const int* __restrict__ bondB,
    float* __restrict__ agg, int E)
{
  __shared__ P1Shared sm;
  const int tid = threadIdx.x;
  for (int i = tid; i < 64*64; i += 128){
    int nn = i >> 6, kk = i & 63;
    sm.Wr1t[i] = (kk < NESF) ? (_Float16)rW1[kk*64 + nn] : (_Float16)0.0f;
    sm.Wr2t[i] = (_Float16)rW2[kk*64 + nn];
  }
  for (int i = tid; i < WNUM*64; i += 128){
    int nn = i >> 6, kk = i & 63;
    sm.Wr3t[i] = (_Float16)rW3[kk*WNUM + nn];
  }
  if (tid < 64){ sm.rb1s[tid] = rb1[tid]; sm.rb2s[tid] = rb2[tid]; }
  for (int i = tid; i < WNUM; i += 128) sm.rb3s[i] = rb3[i];
  __syncthreads();

  const int wave  = tid >> 5;
  const int lane  = tid & 31;
  const int col   = lane & 15;
  const int rbase = (lane >> 4) << 3;   // edge rows owned: 0..7 or 8..15
  auto& W = sm.wv[wave];

  const int tiles = (E + 15) >> 4;
  for (int t = blockIdx.x * 4 + wave; t < tiles; t += gridDim.x * 4){
    const int e0 = t << 4;
    // prefetch next tile's indices into cache (global_prefetch_b8)
    {
      int en = e0 + (gridDim.x << 6);
      if (en < E){ __builtin_prefetch(src + en, 0, 1); __builtin_prefetch(dst + en, 0, 1); }
    }
    if (lane < 16){
      int e = e0 + lane;
      bool valid = (e < E);
      int eC = valid ? e : (E - 1);
      int is = src[eC], id = dst[eC];
      float dirv[3];
      compute_es(W.es + lane*64, x, xA, xB, is, id, bondA[eC], bondB[eC], dirv);
      #pragma unroll
      for (int k = NESF; k < 64; ++k) W.es[lane*64 + k] = (_Float16)0.0f;
      const float* fs = f + is*40;
      #pragma unroll
      for (int u = 0; u < 16; ++u) W.f0s[lane*16 + u] = fs[u];
      #pragma unroll
      for (int u = 0; u < 24; ++u) W.f1s[lane*24 + u] = fs[16 + u];
      #pragma unroll
      for (int u = 0; u < 8; ++u)
        W.pdir[lane*8 + u] = fs[16+u*3]*dirv[0] + fs[17+u*3]*dirv[1] + fs[18+u*3]*dirv[2];
      W.dir[lane*3+0] = dirv[0]; W.dir[lane*3+1] = dirv[1]; W.dir[lane*3+2] = dirv[2];
      W.dste[lane] = valid ? id : -1;
    }
    __builtin_amdgcn_wave_barrier();

    // layer 1: es(16x64) @ Wr1 -> h1, gelu.  A fragments hoisted (loop invariant).
    {
      v16h a0 = load_a_frag(W.es, 64, 0);
      v16h a1 = load_a_frag(W.es, 64, 32);
      #pragma unroll
      for (int nt = 0; nt < 4; ++nt){
        v8f c = {};
        c = wmma_f16(a0, load_b_frag(sm.Wr1t, 64, nt*16, 0),  c);
        c = wmma_f16(a1, load_b_frag(sm.Wr1t, 64, nt*16, 32), c);
        float bias = sm.rb1s[nt*16 + col];
        #pragma unroll
        for (int i = 0; i < 8; ++i)
          W.h[(rbase+i)*64 + nt*16 + col] = (_Float16)gelu_t(c[i] + bias);
      }
    }
    __builtin_amdgcn_wave_barrier();

    // layer 2: h1 @ Wr2 -> h2 (into es), gelu
    {
      v16h a0 = load_a_frag(W.h, 64, 0);
      v16h a1 = load_a_frag(W.h, 64, 32);
      #pragma unroll
      for (int nt = 0; nt < 4; ++nt){
        v8f c = {};
        c = wmma_f16(a0, load_b_frag(sm.Wr2t, 64, nt*16, 0),  c);
        c = wmma_f16(a1, load_b_frag(sm.Wr2t, 64, nt*16, 32), c);
        float bias = sm.rb2s[nt*16 + col];
        #pragma unroll
        for (int i = 0; i < 8; ++i)
          W.es[(rbase+i)*64 + nt*16 + col] = (_Float16)gelu_t(c[i] + bias);
      }
    }
    __builtin_amdgcn_wave_barrier();

    // layer 3 (64 -> 576) consumed tile-by-tile; A (h2) resident in registers.
    float m0acc[8];
    float m1acc[8][3];
    #pragma unroll
    for (int i = 0; i < 8; ++i){ m0acc[i]=0.f; m1acc[i][0]=0.f; m1acc[i][1]=0.f; m1acc[i][2]=0.f; }

    {
      v16h a0 = load_a_frag(W.es, 64, 0);
      v16h a1 = load_a_frag(W.es, 64, 32);

      // region w00: nt 0..15, u = nt, v = col
      for (int nt = 0; nt < 16; ++nt){
        v8f c = {};
        c = wmma_f16(a0, load_b_frag(sm.Wr3t, 64, nt*16, 0),  c);
        c = wmma_f16(a1, load_b_frag(sm.Wr3t, 64, nt*16, 32), c);
        float bias = sm.rb3s[nt*16 + col];
        #pragma unroll
        for (int i = 0; i < 8; ++i)
          m0acc[i] += W.f0s[(rbase+i)*16 + nt] * (c[i] + bias);
      }
      // region w110: nt 16..23, u = nt-16, v = col
      for (int nt = 16; nt < 24; ++nt){
        v8f c = {};
        c = wmma_f16(a0, load_b_frag(sm.Wr3t, 64, nt*16, 0),  c);
        c = wmma_f16(a1, load_b_frag(sm.Wr3t, 64, nt*16, 32), c);
        float bias = sm.rb3s[nt*16 + col];
        #pragma unroll
        for (int i = 0; i < 8; ++i)
          m0acc[i] += W.pdir[(rbase+i)*8 + (nt-16)] * (c[i] + bias);
      }
      // region w01: nt 24..31, u = (nt-24)*2 + col/8, v = col%8 ; Y1 = sqrt(3)*dir
      {
        float dr[8][3];
        #pragma unroll
        for (int i = 0; i < 8; ++i){
          int e = rbase + i;
          dr[i][0] = W.dir[e*3+0]; dr[i][1] = W.dir[e*3+1]; dr[i][2] = W.dir[e*3+2];
        }
        for (int nt = 24; nt < 32; ++nt){
          v8f c = {};
          c = wmma_f16(a0, load_b_frag(sm.Wr3t, 64, nt*16, 0),  c);
          c = wmma_f16(a1, load_b_frag(sm.Wr3t, 64, nt*16, 32), c);
          float bias = sm.rb3s[nt*16 + col];
          int u = (nt - 24)*2 + (col >> 3);
          #pragma unroll
          for (int i = 0; i < 8; ++i){
            float tt = W.f0s[(rbase+i)*16 + u] * (c[i] + bias) * SQ3C;
            m1acc[i][0] += tt * dr[i][0];
            m1acc[i][1] += tt * dr[i][1];
            m1acc[i][2] += tt * dr[i][2];
          }
        }
      }
      // region w10: nt 32..35, u = (nt-32)*2 + col/8, v = col%8
      for (int nt = 32; nt < 36; ++nt){
        v8f c = {};
        c = wmma_f16(a0, load_b_frag(sm.Wr3t, 64, nt*16, 0),  c);
        c = wmma_f16(a1, load_b_frag(sm.Wr3t, 64, nt*16, 32), c);
        float bias = sm.rb3s[nt*16 + col];
        int u = (nt - 32)*2 + (col >> 3);
        #pragma unroll
        for (int i = 0; i < 8; ++i){
          float wv = c[i] + bias;
          m1acc[i][0] += W.f1s[(rbase+i)*24 + u*3 + 0] * wv;
          m1acc[i][1] += W.f1s[(rbase+i)*24 + u*3 + 1] * wv;
          m1acc[i][2] += W.f1s[(rbase+i)*24 + u*3 + 2] * wv;
        }
      }
    }

    // cross-lane pre-reduction: lanes col and col+8 hold complementary halves of
    // msg1[e][v=col&7]; fold so only low half-lanes issue msg1 atomics.
    #pragma unroll
    for (int i = 0; i < 8; ++i){
      m1acc[i][0] += __shfl_xor(m1acc[i][0], 8, 32);
      m1acc[i][1] += __shfl_xor(m1acc[i][1], 8, 32);
      m1acc[i][2] += __shfl_xor(m1acc[i][2], 8, 32);
    }
    const bool doM1 = (col & 8) == 0;
    const int v = col & 7;
    #pragma unroll
    for (int i = 0; i < 8; ++i){
      int e = rbase + i;
      int d = W.dste[e];
      if (d >= 0){
        atomicAdd(agg + d*40 + col, m0acc[i] * INV_FAN);
        if (doM1){
          atomicAdd(agg + d*40 + 16 + v*3 + 0, m1acc[i][0] * INV_FAN);
          atomicAdd(agg + d*40 + 16 + v*3 + 1, m1acc[i][1] * INV_FAN);
          atomicAdd(agg + d*40 + 16 + v*3 + 2, m1acc[i][2] * INV_FAN);
        }
      }
    }
    __builtin_amdgcn_wave_barrier();
  }
}

// ================= Node pass 1: self-interaction, delta_x, sf ==================
__global__ __launch_bounds__(256) void node_pass1(
    const float* __restrict__ f, const float* __restrict__ agg,
    const float* __restrict__ siW0, const float* __restrict__ siW1,
    const float* __restrict__ srW0, const float* __restrict__ chW1,
    float* __restrict__ fnew, float* __restrict__ sf, float* __restrict__ dxw, int n)
{
  __shared__ float sW0[256], sW1[64], sR0[256], sC[8];
  int tid = threadIdx.x;
  if (tid < 256){ sW0[tid] = siW0[tid]; sR0[tid] = srW0[tid]; }
  if (tid < 64) sW1[tid] = siW1[tid];
  if (tid < 8)  sC[tid]  = chW1[tid];
  __syncthreads();
  int nid = blockIdx.x * 256 + tid;
  if (nid >= n) return;
  const float* fp = f   + nid*40;
  const float* ap = agg + nid*40;
  float f0n[16], f1n[24];
  #pragma unroll
  for (int v = 0; v < 16; ++v){
    float sacc = fp[v];
    #pragma unroll
    for (int u = 0; u < 16; ++u) sacc += ap[u] * sW0[u*16 + v];
    f0n[v] = sacc;
  }
  #pragma unroll
  for (int v = 0; v < 8; ++v)
    #pragma unroll
    for (int m = 0; m < 3; ++m){
      float sacc = fp[16 + v*3 + m];
      #pragma unroll
      for (int u = 0; u < 8; ++u) sacc += ap[16 + u*3 + m] * sW1[u*8 + v];
      f1n[v*3 + m] = sacc;
    }
  #pragma unroll
  for (int v = 0; v < 16; ++v) fnew[nid*40 + v] = f0n[v];
  #pragma unroll
  for (int u = 0; u < 24; ++u) fnew[nid*40 + 16 + u] = f1n[u];
  #pragma unroll
  for (int m = 0; m < 3; ++m){
    float dxv = 0.f;
    #pragma unroll
    for (int u = 0; u < 8; ++u) dxv += f1n[u*3 + m] * sC[u];
    dxw[nid*3 + m] = dxv;
  }
  #pragma unroll
  for (int v = 0; v < 16; ++v){
    float sacc = 0.f;
    #pragma unroll
    for (int u = 0; u < 16; ++u) sacc += f0n[u] * sR0[u*16 + v];
    sf[nid*16 + v] = sacc;
  }
}

// ================= Pass 2: edge attention MLP + neighbor scatter ===============
struct P2Shared {
  alignas(16) _Float16 We1t[64*96];     // [n][k], k>=69 zero
  alignas(16) _Float16 We2t[64*64];
  float eb1s[64], eb2s[64], ew3s[64], eb3s;
  struct {
    alignas(16) _Float16 ctx[16*96];    // 37 es | 16 sf_src | 16 sf_dst | pad ; then h2 in cols 0..63
    alignas(16) _Float16 h[16*64];
    float dxs[16*3];
    int   dste[16];
  } wv[8];
};

__global__ __launch_bounds__(256) void edge_pass2(
    const float* __restrict__ x, const float* __restrict__ xA, const float* __restrict__ xB,
    const float* __restrict__ sf,
    const float* __restrict__ eW1, const float* __restrict__ eb1,
    const float* __restrict__ eW2, const float* __restrict__ eb2,
    const float* __restrict__ eW3, const float* __restrict__ eb3,
    const int* __restrict__ src, const int* __restrict__ dst,
    const int* __restrict__ bondA, const int* __restrict__ bondB,
    float* __restrict__ nbr, int E)
{
  __shared__ P2Shared sm;
  const int tid = threadIdx.x;
  for (int i = tid; i < 64*96; i += 256){
    int nn = i / 96, kk = i - nn*96;
    sm.We1t[i] = (kk < 69) ? (_Float16)eW1[kk*64 + nn] : (_Float16)0.0f;
  }
  for (int i = tid; i < 64*64; i += 256){
    int nn = i >> 6, kk = i & 63;
    sm.We2t[i] = (_Float16)eW2[kk*64 + nn];
  }
  if (tid < 64){ sm.eb1s[tid] = eb1[tid]; sm.eb2s[tid] = eb2[tid]; sm.ew3s[tid] = eW3[tid]; }
  if (tid == 0) sm.eb3s = eb3[0];
  __syncthreads();

  const int wave  = tid >> 5;
  const int lane  = tid & 31;
  const int col   = lane & 15;
  const int rbase = (lane >> 4) << 3;
  auto& W = sm.wv[wave];

  const int tiles = (E + 15) >> 4;
  for (int t = blockIdx.x * 8 + wave; t < tiles; t += gridDim.x * 8){
    const int e0 = t << 4;
    {
      int en = e0 + (gridDim.x << 7);
      if (en < E){ __builtin_prefetch(src + en, 0, 1); __builtin_prefetch(dst + en, 0, 1); }
    }
    if (lane < 16){
      int e = e0 + lane;
      bool valid = (e < E);
      int eC = valid ? e : (E - 1);
      int is = src[eC], id = dst[eC];
      float dirv[3];
      _Float16* row = W.ctx + lane*96;
      compute_es(row, x, xA, xB, is, id, bondA[eC], bondB[eC], dirv);
      #pragma unroll
      for (int j = 0; j < 16; ++j){
        row[37 + j] = (_Float16)sf[is*16 + j];
        row[53 + j] = (_Float16)sf[id*16 + j];
      }
      #pragma unroll
      for (int k = 69; k < 96; ++k) row[k] = (_Float16)0.0f;
      W.dxs[lane*3+0] = x[is*3+0] - x[id*3+0];   // dx_to_src = x[src]-x[dst]
      W.dxs[lane*3+1] = x[is*3+1] - x[id*3+1];
      W.dxs[lane*3+2] = x[is*3+2] - x[id*3+2];
      W.dste[lane] = valid ? id : -1;
    }
    __builtin_amdgcn_wave_barrier();

    {
      v16h a0 = load_a_frag(W.ctx, 96, 0);
      v16h a1 = load_a_frag(W.ctx, 96, 32);
      v16h a2 = load_a_frag(W.ctx, 96, 64);
      #pragma unroll
      for (int nt = 0; nt < 4; ++nt){
        v8f c = {};
        c = wmma_f16(a0, load_b_frag(sm.We1t, 96, nt*16, 0),  c);
        c = wmma_f16(a1, load_b_frag(sm.We1t, 96, nt*16, 32), c);
        c = wmma_f16(a2, load_b_frag(sm.We1t, 96, nt*16, 64), c);
        float bias = sm.eb1s[nt*16 + col];
        #pragma unroll
        for (int i = 0; i < 8; ++i)
          W.h[(rbase+i)*64 + nt*16 + col] = (_Float16)gelu_t(c[i] + bias);
      }
    }
    __builtin_amdgcn_wave_barrier();

    {
      v16h a0 = load_a_frag(W.h, 64, 0);
      v16h a1 = load_a_frag(W.h, 64, 32);
      #pragma unroll
      for (int nt = 0; nt < 4; ++nt){
        v8f c = {};
        c = wmma_f16(a0, load_b_frag(sm.We2t, 64, nt*16, 0),  c);
        c = wmma_f16(a1, load_b_frag(sm.We2t, 64, nt*16, 32), c);
        float bias = sm.eb2s[nt*16 + col];
        #pragma unroll
        for (int i = 0; i < 8; ++i)
          W.ctx[(rbase+i)*96 + nt*16 + col] = (_Float16)gelu_t(c[i] + bias);
      }
    }
    __builtin_amdgcn_wave_barrier();

    if (lane < 16){
      float alpha = sm.eb3s;
      #pragma unroll
      for (int k = 0; k < 64; ++k) alpha += (float)W.ctx[lane*96 + k] * sm.ew3s[k];
      int d = W.dste[lane];
      if (d >= 0){
        atomicAdd(nbr + d*3 + 0, alpha * W.dxs[lane*3+0]);
        atomicAdd(nbr + d*3 + 1, alpha * W.dxs[lane*3+1]);
        atomicAdd(nbr + d*3 + 2, alpha * W.dxs[lane*3+2]);
      }
    }
    __builtin_amdgcn_wave_barrier();
  }
}

// ================= Node pass 2: gate MLP + coordinate update ===================
__global__ __launch_bounds__(256) void node_pass2(
    const float* __restrict__ x, const float* __restrict__ xA, const float* __restrict__ xB,
    const float* __restrict__ s, const float* __restrict__ sf,
    const float* __restrict__ dxw, const float* __restrict__ nbr,
    const float* __restrict__ gW1, const float* __restrict__ gb1,
    const float* __restrict__ gW2, const float* __restrict__ gb2,
    const float* __restrict__ gW3, const float* __restrict__ gb3,
    float* __restrict__ xnew, int n)
{
  __shared__ float sg1[16*64];
  __shared__ float sg2[64*64];
  __shared__ float sg3[64*4];
  __shared__ float sb1[64], sb2[64], sb3[4];
  int tid = threadIdx.x;
  for (int i = tid; i < 16*64; i += 256) sg1[i] = gW1[i];
  for (int i = tid; i < 64*64; i += 256) sg2[i] = gW2[i];
  for (int i = tid; i < 64*4;  i += 256) sg3[i] = gW3[i];
  if (tid < 64){ sb1[tid] = gb1[tid]; sb2[tid] = gb2[tid]; }
  if (tid < 4) sb3[tid] = gb3[tid];
  __syncthreads();
  int nid = blockIdx.x * 256 + tid;
  if (nid >= n) return;
  float sfl[16];
  #pragma unroll
  for (int u = 0; u < 16; ++u) sfl[u] = sf[nid*16 + u];
  float h1[64];
  for (int j = 0; j < 64; ++j){
    float acc = sb1[j];
    #pragma unroll
    for (int u = 0; u < 16; ++u) acc += sfl[u] * sg1[u*64 + j];
    h1[j] = gelu_t(acc);
  }
  float g4[4] = { sb3[0], sb3[1], sb3[2], sb3[3] };
  for (int j = 0; j < 64; ++j){
    float acc = sb2[j];
    for (int u = 0; u < 64; ++u) acc += h1[u] * sg2[u*64 + j];
    acc = gelu_t(acc);
    #pragma unroll
    for (int c = 0; c < 4; ++c) g4[c] += acc * sg3[j*4 + c];
  }
  float sv = s[nid];
  #pragma unroll
  for (int m = 0; m < 3; ++m){
    float xv = x[nid*3 + m];
    float xn = xv + g4[0] * dxw[nid*3 + m]
                  + g4[1] * nbr[nid*3 + m]
                  + g4[2] * (1.0f - sv) * (xA[nid*3 + m] - xv)
                  + g4[3] * sv * (xB[nid*3 + m] - xv);
    xnew[nid*3 + m] = xn;
  }
}

__global__ void zero_ws(float* p, int n){
  int i = blockIdx.x * blockDim.x + threadIdx.x;
  if (i < n) p[i] = 0.0f;
}

extern "C" void kernel_launch(void* const* d_in, const int* in_sizes, int n_in,
                              void* d_out, int out_size, void* d_ws, size_t ws_size,
                              hipStream_t stream) {
  const float* f    = (const float*)d_in[0];
  const float* x    = (const float*)d_in[1];
  const float* xA   = (const float*)d_in[2];
  const float* xB   = (const float*)d_in[3];
  const float* s    = (const float*)d_in[4];
  const float* rW1  = (const float*)d_in[5];  const float* rb1 = (const float*)d_in[6];
  const float* rW2  = (const float*)d_in[7];  const float* rb2 = (const float*)d_in[8];
  const float* rW3  = (const float*)d_in[9];  const float* rb3 = (const float*)d_in[10];
  const float* eW1  = (const float*)d_in[11]; const float* eb1 = (const float*)d_in[12];
  const float* eW2  = (const float*)d_in[13]; const float* eb2 = (const float*)d_in[14];
  const float* eW3  = (const float*)d_in[15]; const float* eb3 = (const float*)d_in[16];
  const float* gW1  = (const float*)d_in[17]; const float* gb1 = (const float*)d_in[18];
  const float* gW2  = (const float*)d_in[19]; const float* gb2 = (const float*)d_in[20];
  const float* gW3  = (const float*)d_in[21]; const float* gb3 = (const float*)d_in[22];
  const float* siW0 = (const float*)d_in[23];
  const float* siW1 = (const float*)d_in[24];
  const float* srW0 = (const float*)d_in[25];
  const float* chW1 = (const float*)d_in[26];
  const int* src   = (const int*)d_in[27];
  const int* dst   = (const int*)d_in[28];
  const int* bondA = (const int*)d_in[29];
  const int* bondB = (const int*)d_in[30];

  const int E = in_sizes[27];
  const int n = in_sizes[1] / 3;

  float* ws   = (float*)d_ws;
  float* agg  = ws;                 // n*40
  float* sf   = agg + (size_t)n*40; // n*16
  float* dxw  = sf  + (size_t)n*16; // n*3
  float* nbr  = dxw + (size_t)n*3;  // n*3
  float* fnew = (float*)d_out;          // n*40
  float* xnew = fnew + (size_t)n*40;    // n*3

  int zn = n*40 + n*16 + n*3 + n*3;
  zero_ws<<<(zn + 255)/256, 256, 0, stream>>>(ws, zn);

  int tiles = (E + 15) / 16;
  int g1 = imin_h((tiles + 3) / 4, 4096);
  edge_pass1<<<g1, 128, 0, stream>>>(f, x, xA, xB, rW1, rb1, rW2, rb2, rW3, rb3,
                                     src, dst, bondA, bondB, agg, E);
  node_pass1<<<(n + 255)/256, 256, 0, stream>>>(f, agg, siW0, siW1, srW0, chW1,
                                                fnew, sf, dxw, n);
  int g2 = imin_h((tiles + 7) / 8, 4096);
  edge_pass2<<<g2, 256, 0, stream>>>(x, xA, xB, sf, eW1, eb1, eW2, eb2, eW3, eb3,
                                     src, dst, bondA, bondB, nbr, E);
  node_pass2<<<(n + 255)/256, 256, 0, stream>>>(x, xA, xB, s, sf, dxw, nbr,
                                                gW1, gb1, gW2, gb2, gW3, gb3, xnew, n);
}
